// SparseAutoencoder_28656021799341
// MI455X (gfx1250) — matrix operand order repaired and checked by reading, not compiled, not run on hardware
//
#include <hip/hip_runtime.h>
#include <stdint.h>

// ---------------------------------------------------------------- types
typedef __attribute__((ext_vector_type(16))) __bf16 v16bf;
typedef __attribute__((ext_vector_type(8)))  __bf16 v8bf;
typedef __attribute__((ext_vector_type(8)))  float  v8f;
typedef __attribute__((ext_vector_type(4)))  float  v4f;
typedef __attribute__((ext_vector_type(4)))  int    v4i;

#define N_ROWS 32768          // B*L = 8*4096
#define D_DIM  768
#define H_DIM  256
#define TOPK_K 50
#define BPAD   40             // padded LDS row stride in bf16 (80B)

#if defined(__HIP_DEVICE_COMPILE__) && \
    __has_builtin(__builtin_amdgcn_global_load_async_to_lds_b128) && \
    __has_builtin(__builtin_amdgcn_s_wait_asynccnt)
#define SAE_ASYNC 1
#else
#define SAE_ASYNC 0
#endif

#if SAE_ASYNC
__device__ __forceinline__ void sae_async_b128(const void* g, void* l) {
  // async copy 16B global -> LDS, tracked by ASYNCcnt
  __builtin_amdgcn_global_load_async_to_lds_b128((v4i*)g, (v4i*)l, 0, 0);
}
__device__ __forceinline__ void sae_wait_async() {
  __builtin_amdgcn_s_wait_asynccnt(0);
}
#endif

__device__ __forceinline__ v16bf sae_lds_bfrag(const __bf16* p) {
  // 32B LDS read as two 16B-aligned halves (row stride 80B is not 32B aligned)
  v8bf lo = *(const v8bf*)(p);
  v8bf hi = *(const v8bf*)(p + 8);
  return __builtin_shufflevector(lo, hi, 0,1,2,3,4,5,6,7,8,9,10,11,12,13,14,15);
}

// ---------------------------------------------------------------- prep:
// W = prev + alpha*(dec - prev)  -> d_out W section (f32)
// also bf16 copies of W and enc_W into workspace for the WMMA GEMMs.
__global__ __launch_bounds__(256) void sae_prep_kernel(
    const float* __restrict__ encW, const float* __restrict__ decW,
    const float* __restrict__ prevW, const float* __restrict__ alpha_p,
    float* __restrict__ W_out, __bf16* __restrict__ encWb,
    __bf16* __restrict__ Wb) {
  int i = blockIdx.x * 256 + threadIdx.x;   // 0 .. D*H-1 (== H*D)
  if (i < D_DIM * H_DIM) {
    float a = alpha_p[0];
    float w = prevW[i] + a * (decW[i] - prevW[i]);
    W_out[i] = w;
    Wb[i]    = (__bf16)w;
    encWb[i] = (__bf16)encW[i];
  }
}

// ---------------------------------------------------------------- encoder:
// latent[N,H] = x[N,D] @ enc_W[H,D]^T + b    (bf16 WMMA, f32 accumulate)
// WG = 8 waves; wave: 16 rows x 256 cols (16 accum tiles). B tile (256x32 bf16)
// is async-copied to LDS once per WG per k-step, double-buffered.
__global__ __launch_bounds__(256) void sae_enc_kernel(
    const float* __restrict__ x, const __bf16* __restrict__ encWb,
    const float* __restrict__ encb, float* __restrict__ latent) {
  const int tid  = threadIdx.x;
  const int wave = tid >> 5;
  const int lane = tid & 31;
  const int lmod = lane & 15;
  const int lhi  = lane >> 4;                 // 0 or 1 (half-wave)
  const int mbase = blockIdx.x * 128 + wave * 16;

  const float* xrow = x + (size_t)(mbase + lmod) * D_DIM;

  v8f acc[16];
#pragma unroll
  for (int t = 0; t < 16; ++t) acc[t] = (v8f){0.f,0.f,0.f,0.f,0.f,0.f,0.f,0.f};

#if SAE_ASYNC
  __shared__ __bf16 sB[2][256 * BPAD];        // 2 x 20 KB, padded rows
  // stage k-tile 0: thread t copies row t (32 bf16 = 4 x 16B chunks)
  {
    const __bf16* g = encWb + (size_t)tid * D_DIM;
    __bf16* l = &sB[0][tid * BPAD];
#pragma unroll
    for (int c = 0; c < 4; ++c) sae_async_b128(g + c * 8, l + c * 8);
  }
  for (int kt = 0; kt < D_DIM / 32; ++kt) {
    const int cur = kt & 1;
    sae_wait_async();
    __syncthreads();                          // sB[cur] ready, sB[cur^1] free
    if (kt + 1 < D_DIM / 32) {                // prefetch next tile during WMMAs
      const __bf16* g = encWb + (size_t)tid * D_DIM + (kt + 1) * 32;
      __bf16* l = &sB[cur ^ 1][tid * BPAD];
#pragma unroll
      for (int c = 0; c < 4; ++c) sae_async_b128(g + c * 8, l + c * 8);
    }
    // A fragment: lane holds row M=lane%16, K = k0..k0+15
    const int k0 = kt * 32 + lhi * 16;
    v4f a0 = *(const v4f*)(xrow + k0);
    v4f a1 = *(const v4f*)(xrow + k0 + 4);
    v4f a2 = *(const v4f*)(xrow + k0 + 8);
    v4f a3 = *(const v4f*)(xrow + k0 + 12);
    v16bf afrag;
#pragma unroll
    for (int j = 0; j < 4; ++j) {
      afrag[j]      = (__bf16)a0[j];
      afrag[4 + j]  = (__bf16)a1[j];
      afrag[8 + j]  = (__bf16)a2[j];
      afrag[12 + j] = (__bf16)a3[j];
    }
#pragma unroll
    for (int ht = 0; ht < 16; ++ht) {
      v16bf bfrag = sae_lds_bfrag(&sB[cur][(ht * 16 + lmod) * BPAD + lhi * 16]);
      acc[ht] = __builtin_amdgcn_wmma_f32_16x16x32_bf16(
          false, afrag, false, bfrag, (short)0, acc[ht], false, false);
    }
  }
#else
  const __bf16* brow = encWb + (size_t)lmod * D_DIM;
  for (int kt = 0; kt < D_DIM / 32; ++kt) {
    const int k0 = kt * 32 + lhi * 16;
    v4f a0 = *(const v4f*)(xrow + k0);
    v4f a1 = *(const v4f*)(xrow + k0 + 4);
    v4f a2 = *(const v4f*)(xrow + k0 + 8);
    v4f a3 = *(const v4f*)(xrow + k0 + 12);
    v16bf afrag;
#pragma unroll
    for (int j = 0; j < 4; ++j) {
      afrag[j]      = (__bf16)a0[j];
      afrag[4 + j]  = (__bf16)a1[j];
      afrag[8 + j]  = (__bf16)a2[j];
      afrag[12 + j] = (__bf16)a3[j];
    }
    const int kb = kt * 32 + lhi * 16;
#pragma unroll
    for (int ht = 0; ht < 16; ++ht) {
      v16bf bfrag = *(const v16bf*)(brow + (size_t)ht * 16 * D_DIM + kb);
      acc[ht] = __builtin_amdgcn_wmma_f32_16x16x32_bf16(
          false, afrag, false, bfrag, (short)0, acc[ht], false, false);
    }
  }
#endif
  // C/D layout: VGPR j -> M = j + 8*lhi, N = lane%16
#pragma unroll
  for (int ht = 0; ht < 16; ++ht) {
    const int col  = ht * 16 + lmod;
    const float bias = encb[col];
#pragma unroll
    for (int j = 0; j < 8; ++j) {
      const int row = mbase + j + lhi * 8;
      latent[(size_t)row * H_DIM + col] = acc[ht][j] + bias;
    }
  }
}

// ---------------------------------------------------------------- top-k:
// one wave32 per row; exact 50th-largest |v| via bitwise binary search on
// the float bit pattern (order-preserving for |v| >= 0). Emits 32B bitmask.
__global__ __launch_bounds__(256) void sae_topk_kernel(
    const float* __restrict__ latent, unsigned char* __restrict__ maskb) {
  const int wave = threadIdx.x >> 5;
  const int lane = threadIdx.x & 31;
  const int row  = blockIdx.x * 8 + wave;
  const float* lr = latent + (size_t)row * H_DIM;

  v4f v0 = *(const v4f*)(lr + lane * 8);
  v4f v1 = *(const v4f*)(lr + lane * 8 + 4);
  unsigned u[8];
#pragma unroll
  for (int j = 0; j < 4; ++j) {
    u[j]     = __float_as_uint(v0[j]) & 0x7fffffffu;
    u[4 + j] = __float_as_uint(v1[j]) & 0x7fffffffu;
  }
  unsigned thr = 0u;
  for (int bit = 30; bit >= 0; --bit) {
    const unsigned cand = thr | (1u << bit);
    int c = 0;
#pragma unroll
    for (int j = 0; j < 8; ++j) c += (u[j] >= cand) ? 1 : 0;
#pragma unroll
    for (int off = 16; off > 0; off >>= 1) c += __shfl_xor(c, off, 32);
    if (c >= TOPK_K) thr = cand;
  }
  unsigned m = 0;
#pragma unroll
  for (int j = 0; j < 8; ++j) m |= (u[j] >= thr ? 1u : 0u) << j;
  maskb[(size_t)row * 32 + lane] = (unsigned char)m;
}

// ---------------------------------------------------------------- decoder:
// mod[N,D] = (latent*mask)[N,H] @ W[D,H]^T    (bf16 WMMA, f32 accumulate)
// grid.y: 4 slices of 192 D-cols (12 tiles). B tile (192x32 bf16) async-copied
// to LDS per k-step, double-buffered; latent re-reads hit L2.
__global__ __launch_bounds__(256) void sae_dec_kernel(
    const float* __restrict__ latent, const unsigned char* __restrict__ maskb,
    const __bf16* __restrict__ Wb, float* __restrict__ mod) {
  const int tid  = threadIdx.x;
  const int wave = tid >> 5;
  const int lane = tid & 31;
  const int lmod = lane & 15;
  const int lhi  = lane >> 4;
  const int mbase = blockIdx.x * 128 + wave * 16;
  const int dbase = blockIdx.y * 192;

  const int arow = mbase + lmod;
  const float* lrow = latent + (size_t)arow * H_DIM;
  const unsigned char* mrow = maskb + (size_t)arow * 32;

  v8f acc[12];
#pragma unroll
  for (int t = 0; t < 12; ++t) acc[t] = (v8f){0.f,0.f,0.f,0.f,0.f,0.f,0.f,0.f};

#if SAE_ASYNC
  __shared__ __bf16 sB[2][192 * BPAD];        // 2 x 15 KB
  // stage: 192 rows x 4 chunks = 768 chunks; thread does 3 (chunk = tid+256*i)
  {
#pragma unroll
    for (int i = 0; i < 3; ++i) {
      const int c   = tid + i * 256;
      const int row = c >> 2, ch = c & 3;
      sae_async_b128(Wb + (size_t)(dbase + row) * H_DIM + ch * 8,
                     &sB[0][row * BPAD + ch * 8]);
    }
  }
  for (int kt = 0; kt < H_DIM / 32; ++kt) {
    const int cur = kt & 1;
    sae_wait_async();
    __syncthreads();
    if (kt + 1 < H_DIM / 32) {
#pragma unroll
      for (int i = 0; i < 3; ++i) {
        const int c   = tid + i * 256;
        const int row = c >> 2, ch = c & 3;
        sae_async_b128(Wb + (size_t)(dbase + row) * H_DIM + (kt + 1) * 32 + ch * 8,
                       &sB[cur ^ 1][row * BPAD + ch * 8]);
      }
    }
    const int k0 = kt * 32 + lhi * 16;        // multiple of 8 -> one mask byte
    v4f a0 = *(const v4f*)(lrow + k0);
    v4f a1 = *(const v4f*)(lrow + k0 + 4);
    v4f a2 = *(const v4f*)(lrow + k0 + 8);
    v4f a3 = *(const v4f*)(lrow + k0 + 12);
    const unsigned m0 = mrow[k0 >> 3];
    const unsigned m1 = mrow[(k0 + 8) >> 3];
    v16bf afrag;
#pragma unroll
    for (int j = 0; j < 4; ++j) {
      afrag[j]      = (__bf16)(((m0 >> j)       & 1u) ? a0[j] : 0.f);
      afrag[4 + j]  = (__bf16)(((m0 >> (4 + j)) & 1u) ? a1[j] : 0.f);
      afrag[8 + j]  = (__bf16)(((m1 >> j)       & 1u) ? a2[j] : 0.f);
      afrag[12 + j] = (__bf16)(((m1 >> (4 + j)) & 1u) ? a3[j] : 0.f);
    }
#pragma unroll
    for (int dt = 0; dt < 12; ++dt) {
      v16bf bfrag = sae_lds_bfrag(&sB[cur][(dt * 16 + lmod) * BPAD + lhi * 16]);
      acc[dt] = __builtin_amdgcn_wmma_f32_16x16x32_bf16(
          false, afrag, false, bfrag, (short)0, acc[dt], false, false);
    }
  }
#else
  const __bf16* brow = Wb + (size_t)(dbase + lmod) * H_DIM;
  for (int kt = 0; kt < H_DIM / 32; ++kt) {
    const int k0 = kt * 32 + lhi * 16;
    v4f a0 = *(const v4f*)(lrow + k0);
    v4f a1 = *(const v4f*)(lrow + k0 + 4);
    v4f a2 = *(const v4f*)(lrow + k0 + 8);
    v4f a3 = *(const v4f*)(lrow + k0 + 12);
    const unsigned m0 = mrow[k0 >> 3];
    const unsigned m1 = mrow[(k0 + 8) >> 3];
    v16bf afrag;
#pragma unroll
    for (int j = 0; j < 4; ++j) {
      afrag[j]      = (__bf16)(((m0 >> j)       & 1u) ? a0[j] : 0.f);
      afrag[4 + j]  = (__bf16)(((m0 >> (4 + j)) & 1u) ? a1[j] : 0.f);
      afrag[8 + j]  = (__bf16)(((m1 >> j)       & 1u) ? a2[j] : 0.f);
      afrag[12 + j] = (__bf16)(((m1 >> (4 + j)) & 1u) ? a3[j] : 0.f);
    }
    const int kb = kt * 32 + lhi * 16;
#pragma unroll
    for (int dt = 0; dt < 12; ++dt) {
      v16bf bfrag = *(const v16bf*)(brow + (size_t)dt * 16 * H_DIM + kb);
      acc[dt] = __builtin_amdgcn_wmma_f32_16x16x32_bf16(
          false, afrag, false, bfrag, (short)0, acc[dt], false, false);
    }
  }
#endif
#pragma unroll
  for (int dt = 0; dt < 12; ++dt) {
    const int col = dbase + dt * 16 + lmod;
#pragma unroll
    for (int j = 0; j < 8; ++j) {
      const int row = mbase + j + lhi * 8;
      mod[(size_t)row * D_DIM + col] = acc[dt][j];
    }
  }
}

// ---------------------------------------------------------------- launch
extern "C" void kernel_launch(void* const* d_in, const int* in_sizes, int n_in,
                              void* d_out, int out_size, void* d_ws, size_t ws_size,
                              hipStream_t stream) {
  const float* x     = (const float*)d_in[0];  // [32768, 768]
  const float* encW  = (const float*)d_in[1];  // [256, 768]
  const float* encb  = (const float*)d_in[2];  // [256]
  const float* decW  = (const float*)d_in[3];  // [768, 256]
  const float* prevW = (const float*)d_in[4];  // [768, 256]
  const float* alpha = (const float*)d_in[5];  // [1]

  float* out    = (float*)d_out;
  float* mod    = out;                                       // 32768*768
  float* latent = out + (size_t)N_ROWS * D_DIM;              // 32768*256
  float* W_out  = latent + (size_t)N_ROWS * H_DIM;           // 768*256

  char* ws = (char*)d_ws;
  __bf16* encWb = (__bf16*)ws;                               // 384 KB
  __bf16* Wb    = (__bf16*)(ws + 393216);                    // 384 KB
  unsigned char* maskb = (unsigned char*)(ws + 786432);      // 1 MB

  sae_prep_kernel<<<(D_DIM * H_DIM + 255) / 256, 256, 0, stream>>>(
      encW, decW, prevW, alpha, W_out, encWb, Wb);
  sae_enc_kernel<<<N_ROWS / 128, 256, 0, stream>>>(x, encWb, encb, latent);
  sae_topk_kernel<<<N_ROWS / 8, 256, 0, stream>>>(latent, maskb);
  sae_dec_kernel<<<dim3(N_ROWS / 128, D_DIM / 192), 256, 0, stream>>>(
      latent, maskb, Wb, mod);
}
